// Encoder_75539884802389
// MI455X (gfx1250) — matrix operand order repaired and checked by reading, not compiled, not run on hardware
//
#include <hip/hip_runtime.h>

// ---------------------------------------------------------------------------
// MI455X (gfx1250) implementation of the R-Net style encoder.
// All dense math runs through V_WMMA_F32_16X16X32_F16 (f32->f16 fragment
// conversion on load). Recurrent chains with batch=16 run in a fused
// persistent-LDS kernel (one workgroup per direction, Wh cached in WMMA
// register fragments, h in LDS) to avoid per-timestep launch latency, which
// is the binding constraint for this tiny (~0.25 TFLOP) model.
//
// GEMM contract: M and N are multiples of 16 at every call site (checked by
// construction), so fragment loads are unguarded vector (b128) loads; only
// the K tail (K=556 case) uses branch-free clamped/select loads.
// ---------------------------------------------------------------------------

typedef __attribute__((ext_vector_type(16))) _Float16 v16h;
typedef __attribute__((ext_vector_type(8)))  float    v8f;

#define WMMA_F16(a, b, c) \
  __builtin_amdgcn_wmma_f32_16x16x32_f16(false, (a), false, (b), (short)0, (c), false, false)

__device__ __forceinline__ float sigmf(float x) { return 1.0f / (1.0f + expf(-x)); }

// ---------------------------------------------------------------------------
// Generic WMMA GEMM: C[M,N] = A[M,K] * W[N,K]^T (+ bias[N]).
// One wave per 16x16 output tile; K tiled by 32.
// Fragment layouts used here (contiguous K, same order for A and B; not the ISA table order):
//   A 16x32 f16: lane holds row (lane&15), K block (lane>>4)*16 .. +16 (contiguous)
//   B 32x16 f16: lane holds col (lane&15), K block (lane>>4)*16 .. +16 (contiguous)
//   C/D: lane holds col (lane&15), rows v + 8*(lane>>4)
// Requires: M % 16 == 0, N % 16 == 0, rows 16-byte aligned (lda % 4 == 0).
// ---------------------------------------------------------------------------
__global__ void gemm_wmma_kernel(const float* __restrict__ A, int lda,
                                 const float* __restrict__ W, int ldw,
                                 const float* __restrict__ bias,
                                 float* __restrict__ C, int ldc,
                                 int M, int N, int K) {
  int wave = (int)((blockIdx.x * blockDim.x + threadIdx.x) >> 5);
  int lane = threadIdx.x & 31;
  int ntiles = N >> 4;
  int mtiles = M >> 4;
  if (wave >= ntiles * mtiles) return;
  int nt = wave % ntiles, mt = wave / ntiles;
  int m0 = mt << 4, n0 = nt << 4;
  int ln = lane & 15, hi = lane >> 4;
  const float* Ar = A + (size_t)(m0 + ln) * lda;
  const float* Wr = W + (size_t)(n0 + ln) * ldw;

  v8f c = {};
  int K32 = K & ~31;
  for (int k0 = 0; k0 < K32; k0 += 32) {
    // A fragment: 16 contiguous floats at k0+hi*16 (4x b128 loads)
    float av[16];
    ((float4*)av)[0] = *(const float4*)(Ar + k0 + hi * 16);
    ((float4*)av)[1] = *(const float4*)(Ar + k0 + hi * 16 + 4);
    ((float4*)av)[2] = *(const float4*)(Ar + k0 + hi * 16 + 8);
    ((float4*)av)[3] = *(const float4*)(Ar + k0 + hi * 16 + 12);
    // B fragment: 16 contiguous floats at k0+hi*16 (4x b128 loads)
    float bw[16];
    ((float4*)bw)[0] = *(const float4*)(Wr + k0 + hi * 16);
    ((float4*)bw)[1] = *(const float4*)(Wr + k0 + hi * 16 + 4);
    ((float4*)bw)[2] = *(const float4*)(Wr + k0 + hi * 16 + 8);
    ((float4*)bw)[3] = *(const float4*)(Wr + k0 + hi * 16 + 12);
    v16h a, b;
#pragma unroll
    for (int e = 0; e < 16; ++e) {
      a[e] = (_Float16)av[e];
      b[e] = (_Float16)bw[e];
    }
    c = WMMA_F16(a, b, c);
  }
  if (K32 < K) {
    // Ragged K tail: branch-free (clamped address + value select).
    v16h a, b;
#pragma unroll
    for (int e = 0; e < 8; ++e) {
      int k1 = K32 + hi * 16 + e;
      int k2 = K32 + hi * 16 + 8 + e;
      float v1 = Ar[k1 < K ? k1 : 0];
      float v2 = Ar[k2 < K ? k2 : 0];
      a[e]     = (_Float16)(k1 < K ? v1 : 0.0f);
      a[e + 8] = (_Float16)(k2 < K ? v2 : 0.0f);
    }
#pragma unroll
    for (int e = 0; e < 16; ++e) {
      int k = K32 + hi * 16 + e;
      float v = Wr[k < K ? k : 0];
      b[e] = (_Float16)(k < K ? v : 0.0f);
    }
    c = WMMA_F16(a, b, c);
  }
  int col = n0 + ln;
  float bv = bias ? bias[col] : 0.0f;
#pragma unroll
  for (int v = 0; v < 8; ++v)
    C[(size_t)(m0 + v + 8 * hi) * ldc + col] = c[v] + bv;
}

// ---------------------------------------------------------------------------
// Fused 3-phase GRU sequence kernel for batch == 16 (wave32, 16 waves).
// Per block (one per direction): h[16,256] in LDS (f32), gh[16,768] in LDS
// (f16), Wh pre-converted to 24 register-resident v16h B-fragments per wave.
// Per step: 3 WMMA tile GEMMs from LDS, barrier, fused gate math, barrier.
// ---------------------------------------------------------------------------
struct SeqArgs {
  const float* gi[2];   // [16, T, 768]
  const float* Wh[2];   // [768, 256]
  const float* bh[2];   // [768]
  float*       y[2];    // output base (dir column offset baked in)
  int          back[2];
  int T;
  int y_rstride;        // floats between batch rows
  int y_tstride;        // floats between timesteps
};

__global__ void __launch_bounds__(512) gru_seq16_kernel(SeqArgs args) {
  __shared__ float    hbuf[16 * 256];
  __shared__ _Float16 ghs[16 * 768];
  int d = blockIdx.x;
  const float* gi = args.gi[d];
  const float* Wh = args.Wh[d];
  const float* bh = args.bh[d];
  float*       y  = args.y[d];
  int back = args.back[d];
  int T = args.T;
  int tid = threadIdx.x;
  int wave = tid >> 5, lane = tid & 31;
  int ln = lane & 15, hi = lane >> 4;

  // Preload Wh fragments (reused every timestep) and bias values.
  v16h bf[3][8];
  float bv[3];
#pragma unroll
  for (int tt = 0; tt < 3; ++tt) {
    int wrow = ((wave * 3 + tt) << 4) + ln;
    const float* Wr = Wh + (size_t)wrow * 256;
    bv[tt] = bh[wrow];
#pragma unroll
    for (int kk = 0; kk < 8; ++kk) {
      float bw[16];
      ((float4*)bw)[0] = *(const float4*)(Wr + kk * 32 + hi * 16);
      ((float4*)bw)[1] = *(const float4*)(Wr + kk * 32 + hi * 16 + 4);
      ((float4*)bw)[2] = *(const float4*)(Wr + kk * 32 + hi * 16 + 8);
      ((float4*)bw)[3] = *(const float4*)(Wr + kk * 32 + hi * 16 + 12);
#pragma unroll
      for (int e = 0; e < 16; ++e) bf[tt][kk][e] = (_Float16)bw[e];
    }
  }
  for (int i = tid; i < 16 * 256; i += 512) hbuf[i] = 0.0f;
  __syncthreads();

  for (int step = 0; step < T; ++step) {
    int t = back ? (T - 1 - step) : step;
    // Phase 1: gh = h @ Wh^T  (3 tiles per wave, A-fragments from LDS h)
#pragma unroll
    for (int tt = 0; tt < 3; ++tt) {
      v8f c = {};
      const float* hrow = &hbuf[ln * 256];
#pragma unroll
      for (int kk = 0; kk < 8; ++kk) {
        float av[16];
        ((float4*)av)[0] = *(const float4*)(hrow + kk * 32 + hi * 16);
        ((float4*)av)[1] = *(const float4*)(hrow + kk * 32 + hi * 16 + 4);
        ((float4*)av)[2] = *(const float4*)(hrow + kk * 32 + hi * 16 + 8);
        ((float4*)av)[3] = *(const float4*)(hrow + kk * 32 + hi * 16 + 12);
        v16h a;
#pragma unroll
        for (int e = 0; e < 16; ++e) a[e] = (_Float16)av[e];
        c = WMMA_F16(a, bf[tt][kk], c);
      }
      int col = ((wave * 3 + tt) << 4) + ln;
#pragma unroll
      for (int v = 0; v < 8; ++v)
        ghs[(v + 8 * hi) * 768 + col] = (_Float16)(c[v] + bv[tt]);
    }
    __syncthreads();
    // Phase 2: fused gate math + h update + y store
    for (int e = tid; e < 16 * 256; e += 512) {
      int i = e >> 8, j = e & 255;
      const float* gir = gi + ((size_t)i * T + t) * 768;
      const _Float16* ghr = &ghs[i * 768];
      float r = sigmf(gir[j] + (float)ghr[j]);
      float z = sigmf(gir[256 + j] + (float)ghr[256 + j]);
      float n = tanhf(gir[512 + j] + r * (float)ghr[512 + j]);
      float hn = (1.0f - z) * n + z * hbuf[e];
      hbuf[e] = hn;
      y[(size_t)i * args.y_rstride + (size_t)t * args.y_tstride + j] = hn;
    }
    __syncthreads();
  }
}

// ---------------------------------------------------------------------------
// Elementwise / utility kernels
// ---------------------------------------------------------------------------
__global__ void gru_step_kernel(const float* __restrict__ gi_t, int gi_rstride,
                                const float* __restrict__ gh,
                                float* __restrict__ h,
                                float* __restrict__ y_t, int y_rstride,
                                int Nb) {
  int idx = blockIdx.x * blockDim.x + threadIdx.x;
  if (idx >= Nb * 256) return;
  int i = idx >> 8, j = idx & 255;
  const float* gir = gi_t + (size_t)i * gi_rstride;
  const float* ghr = gh + (size_t)i * 768;
  float r = sigmf(gir[j] + ghr[j]);
  float z = sigmf(gir[256 + j] + ghr[256 + j]);
  float n = tanhf(gir[512 + j] + r * ghr[512 + j]);
  float hn = (1.0f - z) * n + z * h[idx];
  h[idx] = hn;
  y_t[(size_t)i * y_rstride + j] = hn;
}

__global__ void zero_kernel(float* __restrict__ p, int n) {
  int i = blockIdx.x * blockDim.x + threadIdx.x;
  if (i < n) p[i] = 0.0f;
}

__global__ void gather_kernel(const float* __restrict__ table, const int* __restrict__ idx,
                              float* __restrict__ dst, int rows, int cols, int dst_stride) {
  long long i = (long long)blockIdx.x * blockDim.x + threadIdx.x;
  if (i >= (long long)rows * cols) return;
  int r = (int)(i / cols), c = (int)(i % cols);
  dst[(size_t)r * dst_stride + c] = table[(size_t)idx[r] * cols + c];
}

__global__ void copy_strided_kernel(const float* __restrict__ src, int sstride,
                                    float* __restrict__ dst, int dstride,
                                    int rows, int cols) {
  long long i = (long long)blockIdx.x * blockDim.x + threadIdx.x;
  if (i >= (long long)rows * cols) return;
  int r = (int)(i / cols), c = (int)(i % cols);
  dst[(size_t)r * dstride + c] = src[(size_t)r * sstride + c];
}

__global__ void add_kernel(const float* __restrict__ a, const float* __restrict__ b,
                           float* __restrict__ c, int n) {
  int i = blockIdx.x * blockDim.x + threadIdx.x;
  if (i < n) c[i] = a[i] + b[i];
}

// x = sigmoid(gpre) * u   (written in place of gpre)
__global__ void gate_kernel(float* __restrict__ gpre, const float* __restrict__ u, int n) {
  int i = blockIdx.x * blockDim.x + threadIdx.x;
  if (i < n) gpre[i] = sigmf(gpre[i]) * u[i];
}

// s[b,q,k] = sum_h tanh(kp[b,k,h] + qp[b,q,h]) * wv[h]  (one wave per element)
__global__ void attn_score_kernel(const float* __restrict__ kp, const float* __restrict__ qp,
                                  const float* __restrict__ wv, float* __restrict__ s,
                                  int Bn, int Sq, int Sk) {
  int idx = (int)((blockIdx.x * blockDim.x + threadIdx.x) >> 5);
  int lane = threadIdx.x & 31;
  if (idx >= Bn * Sq * Sk) return;
  int k = idx % Sk, t = idx / Sk;
  int q = t % Sq, b = t / Sq;
  const float* kpr = kp + ((size_t)b * Sk + k) * 256;
  const float* qpr = qp + ((size_t)b * Sq + q) * 256;
  float acc = 0.0f;
  for (int h = lane; h < 256; h += 32) acc += tanhf(kpr[h] + qpr[h]) * wv[h];
  for (int off = 16; off > 0; off >>= 1) acc += __shfl_down(acc, off, 32);
  if (lane == 0) s[idx] = acc;
}

// att[b,q,k] = exp(s) / sum_{b',k'} exp(s[b',q,k'])   (faithful global-per-q sum)
__global__ void attn_norm_kernel(const float* __restrict__ s, float* __restrict__ att,
                                 int Bn, int Sq, int Sk) {
  __shared__ float red[256];
  int q = blockIdx.x;
  int tot = Bn * Sk;
  float acc = 0.0f;
  for (int i = threadIdx.x; i < tot; i += blockDim.x) {
    int b = i / Sk, k = i % Sk;
    acc += expf(s[((size_t)b * Sq + q) * Sk + k]);
  }
  red[threadIdx.x] = acc;
  __syncthreads();
  for (int st = blockDim.x >> 1; st > 0; st >>= 1) {
    if ((int)threadIdx.x < st) red[threadIdx.x] += red[threadIdx.x + st];
    __syncthreads();
  }
  float denom = red[0];
  for (int i = threadIdx.x; i < tot; i += blockDim.x) {
    int b = i / Sk, k = i % Sk;
    size_t off = ((size_t)b * Sq + q) * Sk + k;
    att[off] = expf(s[off]) / denom;
  }
}

// u[b,q,h] = sum_k att[b,q,k]*keys[b,k,h]  (+ queries[b,q,h] if given)
__global__ void attn_apply_kernel(const float* __restrict__ att, const float* __restrict__ keys,
                                  const float* __restrict__ queries, float* __restrict__ u,
                                  int Bn, int Sq, int Sk) {
  long long idx = (long long)blockIdx.x * blockDim.x + threadIdx.x;
  if (idx >= (long long)Bn * Sq * 256) return;
  int h = (int)(idx & 255);
  long long t = idx >> 8;
  int q = (int)(t % Sq), b = (int)(t / Sq);
  const float* ar = att + ((size_t)b * Sq + q) * Sk;
  float acc = 0.0f;
  for (int k = 0; k < Sk; ++k) acc += ar[k] * keys[((size_t)b * Sk + k) * 256 + h];
  u[idx] = acc + (queries ? queries[idx] : 0.0f);
}

// logits + softmax head: out[b,:] = softmax(tanh(x[b,:]) @ wv^T)
__global__ void head_kernel(const float* __restrict__ x, const float* __restrict__ wv,
                            float* __restrict__ out) {
  int b = threadIdx.x;
  if (b >= 16) return;
  float l0 = 0.0f, l1 = 0.0f;
  for (int h = 0; h < 256; ++h) {
    float t = tanhf(x[b * 256 + h]);
    l0 += t * wv[h];
    l1 += t * wv[256 + h];
  }
  float mx = fmaxf(l0, l1);
  float e0 = expf(l0 - mx), e1 = expf(l1 - mx);
  out[b * 2 + 0] = e0 / (e0 + e1);
  out[b * 2 + 1] = e1 / (e0 + e1);
}

// ---------------------------------------------------------------------------
// Host-side orchestration
// ---------------------------------------------------------------------------
namespace {

struct Alloc {
  char* base;
  size_t off;
  float* alloc(size_t nfloats) {
    float* p = (float*)(base + off);
    off += ((nfloats * sizeof(float)) + 255) & ~(size_t)255;
    return p;
  }
};

struct Dir { const float *Wi, *Wh, *bi, *bh; };
struct Match { const float *Wk, *Wq, *wv, *Wg, *bg; Dir gru[3]; };

inline void gemm(hipStream_t s, const float* A, int lda, const float* W, int ldw,
                 const float* bias, float* C, int ldc, int M, int N, int K) {
  int tiles = (M / 16) * (N / 16);
  int blocks = (tiles + 3) / 4;  // 4 waves per 128-thread block
  gemm_wmma_kernel<<<blocks, 128, 0, s>>>(A, lda, W, ldw, bias, C, ldc, M, N, K);
}

inline void launch_zero(hipStream_t s, float* p, int n) {
  zero_kernel<<<(n + 255) / 256, 256, 0, s>>>(p, n);
}

// Per-step GRU direction for large batch (char path).  y element (i,t,j) at
// ybase + i*y_rstride + t*y_tstride + j.
void gru_dir_steps(hipStream_t s, Alloc& al, const float* x, int in, int Nb, int T,
                   const Dir& d, float* ybase, int y_rstride, int y_tstride, bool back) {
  size_t mark = al.off;
  float* h   = al.alloc((size_t)Nb * 256);
  float* gh  = al.alloc((size_t)Nb * 768);
  float* git = al.alloc((size_t)Nb * 768);
  launch_zero(s, h, Nb * 256);
  for (int stp = 0; stp < T; ++stp) {
    int t = back ? (T - 1 - stp) : stp;
    gemm(s, x + (size_t)t * in, T * in, d.Wi, in, d.bi, git, 768, Nb, 768, in);
    gemm(s, h, 256, d.Wh, 256, d.bh, gh, 768, Nb, 768, 256);
    gru_step_kernel<<<(Nb * 256 + 255) / 256, 256, 0, s>>>(
        git, 768, gh, h, ybase + (size_t)t * y_tstride, y_rstride, Nb);
  }
  al.off = mark;
}

// 3-layer BiGRU + last-timestep Linear for the char path (Nb in {512,4096}).
void char_bigru(hipStream_t s, Alloc& al, const float* ce_in, int Nb, Dir gru[3][2],
                const float* linW, const float* linb, float* out /*[Nb,256]*/) {
  const int T = 8;
  size_t mark = al.off;
  float* xa = al.alloc((size_t)Nb * T * 512);
  float* xb = al.alloc((size_t)Nb * T * 512);
  const float* x = ce_in;
  int in = 256;
  float* cur = xa;
  for (int l = 0; l < 3; ++l) {
    for (int d2 = 0; d2 < 2; ++d2)
      gru_dir_steps(s, al, x, in, Nb, T, gru[l][d2], cur + d2 * 256, T * 512, 512, d2 == 1);
    x = cur;
    in = 512;
    cur = (cur == xa) ? xb : xa;
  }
  // Linear(2H->H) on last timestep only (reference takes [:, -1, :] afterwards)
  gemm(s, x + (size_t)(T - 1) * 512, T * 512, linW, 512, linb, out, 256, Nb, 256, 512);
  al.off = mark;
}

// 3-layer BiGRU (batch 16, fused sequence kernel) + Linear on all timesteps.
void wc_bigru(hipStream_t s, Alloc& al, const float* cat, int inc, int T,
              Dir gru[3][2], const float* linW, const float* linb,
              float* out /*[16*T,256]*/) {
  size_t mark = al.off;
  float* xa  = al.alloc((size_t)16 * T * 512);
  float* xb  = al.alloc((size_t)16 * T * 512);
  float* gi0 = al.alloc((size_t)16 * T * 768);
  float* gi1 = al.alloc((size_t)16 * T * 768);
  const float* x = cat;
  int in = inc;
  float* cur = xa;
  for (int l = 0; l < 3; ++l) {
    gemm(s, x, in, gru[l][0].Wi, in, gru[l][0].bi, gi0, 768, 16 * T, 768, in);
    gemm(s, x, in, gru[l][1].Wi, in, gru[l][1].bi, gi1, 768, 16 * T, 768, in);
    SeqArgs sa{};
    sa.gi[0] = gi0; sa.Wh[0] = gru[l][0].Wh; sa.bh[0] = gru[l][0].bh; sa.y[0] = cur;       sa.back[0] = 0;
    sa.gi[1] = gi1; sa.Wh[1] = gru[l][1].Wh; sa.bh[1] = gru[l][1].bh; sa.y[1] = cur + 256; sa.back[1] = 1;
    sa.T = T; sa.y_rstride = T * 512; sa.y_tstride = 512;
    gru_seq16_kernel<<<2, 512, 0, s>>>(sa);
    x = cur;
    in = 512;
    cur = (cur == xa) ? xb : xa;
  }
  gemm(s, x, 512, linW, 512, linb, out, 256, 16 * T, 256, 512);
  al.off = mark;
}

// 3-layer unidirectional GRU (batch 16, fused sequence kernel).
void run_gru16(hipStream_t s, Alloc& al, const float* x0, int T, const Dir gru[3],
               float* out /*[16,T,256]*/) {
  size_t mark = al.off;
  float* ya = al.alloc((size_t)16 * T * 256);
  float* yb = al.alloc((size_t)16 * T * 256);
  float* gi = al.alloc((size_t)16 * T * 768);
  float* ybufs[3] = {ya, yb, out};
  const float* x = x0;
  for (int l = 0; l < 3; ++l) {
    gemm(s, x, 256, gru[l].Wi, 256, gru[l].bi, gi, 768, 16 * T, 768, 256);
    SeqArgs sa{};
    sa.gi[0] = gi; sa.Wh[0] = gru[l].Wh; sa.bh[0] = gru[l].bh; sa.y[0] = ybufs[l]; sa.back[0] = 0;
    sa.T = T; sa.y_rstride = T * 256; sa.y_tstride = 256;
    gru_seq16_kernel<<<1, 512, 0, s>>>(sa);
    x = ybufs[l];
  }
  al.off = mark;
}

void match_gated(hipStream_t s, Alloc& al, const float* keys, int Sk,
                 const float* queries, int Sq, const Match& m, float* out) {
  size_t mark = al.off;
  float* kp = al.alloc((size_t)16 * Sk * 256);
  float* qp = al.alloc((size_t)16 * Sq * 256);
  float* sc = al.alloc((size_t)16 * Sq * Sk);
  float* uc = al.alloc((size_t)16 * Sq * 256);
  float* gp = al.alloc((size_t)16 * Sq * 256);
  gemm(s, keys, 256, m.Wk, 256, nullptr, kp, 256, 16 * Sk, 256, 256);
  gemm(s, queries, 256, m.Wq, 256, nullptr, qp, 256, 16 * Sq, 256, 256);
  int tot = 16 * Sq * Sk;
  attn_score_kernel<<<(tot + 3) / 4, 128, 0, s>>>(kp, qp, m.wv, sc, 16, Sq, Sk);
  attn_norm_kernel<<<Sq, 256, 0, s>>>(sc, sc, 16, Sq, Sk);
  int nel = 16 * Sq * 256;
  attn_apply_kernel<<<(nel + 255) / 256, 256, 0, s>>>(sc, keys, queries, uc, 16, Sq, Sk);
  gemm(s, uc, 256, m.Wg, 256, m.bg, gp, 256, 16 * Sq, 256, 256);
  gate_kernel<<<(nel + 255) / 256, 256, 0, s>>>(gp, uc, nel);
  run_gru16(s, al, gp, Sq, m.gru, out);
  al.off = mark;
}

void match_plain(hipStream_t s, Alloc& al, const float* keys, int Sk,
                 const float* query /*[16,256]*/, const Match& m, float* out /*[16,256]*/) {
  size_t mark = al.off;
  float* kp = al.alloc((size_t)16 * Sk * 256);
  float* qp = al.alloc((size_t)16 * 256);
  float* sc = al.alloc((size_t)16 * Sk);
  gemm(s, keys, 256, m.Wk, 256, nullptr, kp, 256, 16 * Sk, 256, 256);
  gemm(s, query, 256, m.Wq, 256, nullptr, qp, 256, 16, 256, 256);
  int tot = 16 * 1 * Sk;
  attn_score_kernel<<<(tot + 3) / 4, 128, 0, s>>>(kp, qp, m.wv, sc, 16, 1, Sk);
  attn_norm_kernel<<<1, 256, 0, s>>>(sc, sc, 16, 1, Sk);
  int nel = 16 * 256;
  attn_apply_kernel<<<(nel + 255) / 256, 256, 0, s>>>(sc, keys, nullptr, out, 16, 1, Sk);
  al.off = mark;
}

}  // namespace

extern "C" void kernel_launch(void* const* d_in, const int* in_sizes, int n_in,
                              void* d_out, int out_size, void* d_ws, size_t ws_size,
                              hipStream_t stream) {
  (void)in_sizes; (void)n_in; (void)out_size; (void)ws_size;
  // ---- walk inputs in setup_inputs() dict order ----
  int ix = 0;
  auto I = [&](void) { return (const int*)d_in[ix++]; };
  auto F = [&](void) { return (const float*)d_in[ix++]; };
  const int* q_word = I();
  const int* q_char = I();
  const int* p_word = I();
  const int* p_char = I();
  const float* word_emb   = F();
  const float* char_table = F();
  Dir char_gru[3][2];
  for (int l = 0; l < 3; ++l)
    for (int d = 0; d < 2; ++d) char_gru[l][d] = {F(), F(), F(), F()};
  const float* char_lin_W = F();
  const float* char_lin_b = F();
  Dir wc_gru[3][2];
  for (int l = 0; l < 3; ++l)
    for (int d = 0; d < 2; ++d) wc_gru[l][d] = {F(), F(), F(), F()};
  const float* wc_lin_W = F();
  const float* wc_lin_b = F();
  auto read_match = [&](bool gated) {
    Match m{};
    m.Wk = F(); m.Wq = F(); m.wv = F();
    if (gated) {
      m.Wg = F(); m.bg = F();
      for (int l = 0; l < 3; ++l) m.gru[l] = {F(), F(), F(), F()};
    }
    return m;
  };
  Match mg_p  = read_match(true);
  Match mg_q  = read_match(true);
  Match mg_rq = read_match(false);
  Match mg_rp = read_match(false);
  const float* ms_l1_W = F();
  const float* ms_l1_b = F();
  const float* ms_l2_W = F();
  const float* ms_l2_b = F();
  const float* ms_wv   = F();

  const int B = 16, SQ = 32, SP = 256, LC = 8, E = 300;
  hipStream_t s = stream;
  Alloc al{(char*)d_ws, 0};

  // ---- persistent intermediates ----
  float* q_emb  = al.alloc((size_t)B * SQ * 256);
  float* p_emb  = al.alloc((size_t)B * SP * 256);
  float* q_sent = al.alloc((size_t)B * SQ * 256);
  float* p_sent = al.alloc((size_t)B * SP * 256);
  float* ce_q   = al.alloc((size_t)B * SQ * 256);
  float* ce_p   = al.alloc((size_t)B * SP * 256);
  float* q_last = al.alloc((size_t)B * 256);
  float* q_r    = al.alloc((size_t)B * 256);
  float* p_r    = al.alloc((size_t)B * 256);
  float* xsum   = al.alloc((size_t)B * 256);
  float* x1     = al.alloc((size_t)B * 256);
  float* x2     = al.alloc((size_t)B * 256);

  // ---- char embedding paths ----
  {
    size_t mk = al.off;
    int Nb = B * SQ;  // 512
    float* cein = al.alloc((size_t)Nb * LC * 256);
    long long tot = (long long)Nb * LC * 256;
    gather_kernel<<<(int)((tot + 255) / 256), 256, 0, s>>>(char_table, q_char, cein, Nb * LC, 256, 256);
    char_bigru(s, al, cein, Nb, char_gru, char_lin_W, char_lin_b, ce_q);
    al.off = mk;
  }
  {
    size_t mk = al.off;
    int Nb = B * SP;  // 4096
    float* cein = al.alloc((size_t)Nb * LC * 256);
    long long tot = (long long)Nb * LC * 256;
    gather_kernel<<<(int)((tot + 255) / 256), 256, 0, s>>>(char_table, p_char, cein, Nb * LC, 256, 256);
    char_bigru(s, al, cein, Nb, char_gru, char_lin_W, char_lin_b, ce_p);
    al.off = mk;
  }

  // ---- word+char BiGRU paths ----
  {
    size_t mk = al.off;
    int rows = B * SQ;
    float* cat = al.alloc((size_t)rows * (E + 256));
    long long tw = (long long)rows * E;
    gather_kernel<<<(int)((tw + 255) / 256), 256, 0, s>>>(word_emb, q_word, cat, rows, E, E + 256);
    long long tc = (long long)rows * 256;
    copy_strided_kernel<<<(int)((tc + 255) / 256), 256, 0, s>>>(ce_q, 256, cat + E, E + 256, rows, 256);
    wc_bigru(s, al, cat, E + 256, SQ, wc_gru, wc_lin_W, wc_lin_b, q_emb);
    al.off = mk;
  }
  {
    size_t mk = al.off;
    int rows = B * SP;
    float* cat = al.alloc((size_t)rows * (E + 256));
    long long tw = (long long)rows * E;
    gather_kernel<<<(int)((tw + 255) / 256), 256, 0, s>>>(word_emb, p_word, cat, rows, E, E + 256);
    long long tc = (long long)rows * 256;
    copy_strided_kernel<<<(int)((tc + 255) / 256), 256, 0, s>>>(ce_p, 256, cat + E, E + 256, rows, 256);
    wc_bigru(s, al, cat, E + 256, SP, wc_gru, wc_lin_W, wc_lin_b, p_emb);
    al.off = mk;
  }

  // ---- match-GRU attention stages ----
  match_gated(s, al, q_emb, SQ, p_emb, SP, mg_p, p_sent);
  match_gated(s, al, p_emb, SP, q_emb, SQ, mg_q, q_sent);

  copy_strided_kernel<<<(16 * 256 + 255) / 256, 256, 0, s>>>(
      q_sent + (size_t)(SQ - 1) * 256, SQ * 256, q_last, 256, B, 256);
  match_plain(s, al, q_emb, SQ, q_last, mg_rq, q_r);
  match_plain(s, al, p_sent, SP, q_r, mg_rp, p_r);

  // ---- head ----
  add_kernel<<<(B * 256 + 255) / 256, 256, 0, s>>>(q_r, p_r, xsum, B * 256);
  gemm(s, xsum, 256, ms_l1_W, 256, ms_l1_b, x1, 256, B, 256, 256);
  gemm(s, x1, 256, ms_l2_W, 256, ms_l2_b, x2, 256, B, 256, 256);
  head_kernel<<<1, 32, 0, s>>>(x2, ms_wv, (float*)d_out);
}
